// GCNBlock_2491081031682
// MI455X (gfx1250) — matrix operand order repaired and checked by reading, not compiled, not run on hardware
//
#include <hip/hip_runtime.h>
#include <hip/hip_bf16.h>

typedef __attribute__((ext_vector_type(16))) __bf16 v16bf;
typedef __attribute__((ext_vector_type(8)))  __bf16 v8bf;
typedef __attribute__((ext_vector_type(8)))  float  v8f;

#define C 128

// ---------------------------------------------------------------------------
// Small elementwise kernels
// ---------------------------------------------------------------------------
__global__ void k_fill(float* __restrict__ p, float v, long long n) {
    long long i = (long long)blockIdx.x * blockDim.x + threadIdx.x;
    if (i < n) p[i] = v;
}

__global__ void k_degree(const int* __restrict__ dst, float* __restrict__ deg, int E) {
    int e = blockIdx.x * blockDim.x + threadIdx.x;
    if (e < E) atomicAdd(&deg[dst[e]], 1.0f);
}

__global__ void k_rsqrt_inplace(float* __restrict__ p, int N) {
    int i = blockIdx.x * blockDim.x + threadIdx.x;
    if (i < N) p[i] = rsqrtf(p[i]);   // deg >= 1 always (self-loop)
}

// acc[i] = (acc[i] + h[i]*dinv[n]^2 + b[c]) , optional ReLU (in-place)
__global__ void k_finalize(float* __restrict__ acc, const float* __restrict__ h,
                           const float* __restrict__ dinv, const float* __restrict__ b,
                           long long NC, int relu) {
    long long i = (long long)blockIdx.x * blockDim.x + threadIdx.x;
    if (i >= NC) return;
    int n = (int)(i >> 7);
    int c = (int)(i & (C - 1));
    float di = dinv[n];
    float v = acc[i] + h[i] * di * di + b[c];
    if (relu) v = fmaxf(v, 0.0f);
    acc[i] = v;
}

// ---------------------------------------------------------------------------
// Edge scatter: one wave32 per edge, lane handles 4 channels (float4 gather,
// 4x global_atomic_add_f32). Accumulator is L2-resident (51MB < 192MB L2).
// ---------------------------------------------------------------------------
__global__ __launch_bounds__(256) void k_scatter(const float* __restrict__ h,
                                                 const int* __restrict__ src,
                                                 const int* __restrict__ dst,
                                                 const float* __restrict__ dinv,
                                                 float* __restrict__ acc, int E) {
    int e = blockIdx.x * 8 + (threadIdx.x >> 5);
    if (e >= E) return;
    int lane = threadIdx.x & 31;
    int s = src[e];
    int d = dst[e];
    float w = dinv[s] * dinv[d];
    const float4 hv = *(const float4*)(h + (long long)s * C + lane * 4);
    float* out = acc + (long long)d * C + lane * 4;
    atomicAdd(out + 0, hv.x * w);
    atomicAdd(out + 1, hv.y * w);
    atomicAdd(out + 2, hv.z * w);
    atomicAdd(out + 3, hv.w * w);
}

// ---------------------------------------------------------------------------
// WMMA GEMM: H[N x 128] = X[N x 128] @ W[128 x 128]  (bf16 in, f32 accumulate)
// Block: 256 threads = 8 waves. W staged transposed in LDS (32 KB), the
// block's 64-row X tile staged as bf16 in LDS (16 KB). Wave w owns column
// tile [16w, 16w+16); loops 4 row-subtiles x 4 K-steps of 32.
// Tail rows handled by clamped loads + guarded stores (no exec juggling).
// ---------------------------------------------------------------------------
__global__ __launch_bounds__(256) void k_gemm_wmma(const float* __restrict__ X,
                                                   const float* __restrict__ W,
                                                   float* __restrict__ H, int N) {
    __shared__ __bf16 sWt[C * C];    // W transposed: sWt[n*128 + k] = W[k][n]
    __shared__ __bf16 sX[64 * C];    // X tile, bf16, row-major

    const int tid = threadIdx.x;
    const int rowBlock = blockIdx.x * 64;

    // Stage W transposed as bf16 (coalesced global reads, once per block)
    for (int i = tid; i < C * C; i += 256) {
        int k = i >> 7;
        int n = i & (C - 1);
        sWt[n * C + k] = (__bf16)W[i];
    }

    // Stage the 64x128 X tile as bf16; clamp row for the ragged tail.
    for (int i = tid * 8; i < 64 * C; i += 256 * 8) {
        int r  = i >> 7;
        int k  = i & (C - 1);
        int gr = min(rowBlock + r, N - 1);
        const float* xp = X + (long long)gr * C + k;
        float4 f0 = *(const float4*)(xp);
        float4 f1 = *(const float4*)(xp + 4);
        v8bf v;
        v[0] = (__bf16)f0.x; v[1] = (__bf16)f0.y; v[2] = (__bf16)f0.z; v[3] = (__bf16)f0.w;
        v[4] = (__bf16)f1.x; v[5] = (__bf16)f1.y; v[6] = (__bf16)f1.z; v[7] = (__bf16)f1.w;
        *(v8bf*)(sX + i) = v;
    }
    __syncthreads();

    const int wave   = tid >> 5;
    const int lane   = tid & 31;
    const int lhalf  = lane & 15;
    const int kBaseA = (lane < 16) ? 0 : 16;   // A-layout K halves
    const int kBaseB = (lane < 16) ? 0 : 16;   // B-layout K halves
    const int col    = wave * 16 + lhalf;
    const bool full  = (rowBlock + 64) <= N;

    // B fragments for all 4 K-steps: 16 contiguous bf16 per lane from sWt.
    const __bf16* pb = sWt + col * C + kBaseB;
    v16bf bf0 = *(const v16bf*)(pb + 0);
    v16bf bf1 = *(const v16bf*)(pb + 32);
    v16bf bf2 = *(const v16bf*)(pb + 64);
    v16bf bf3 = *(const v16bf*)(pb + 96);

    #pragma unroll
    for (int rs = 0; rs < 4; ++rs) {
        // A fragment: row = rs*16 + lhalf, elems 0..15 -> K=kk+kBaseA+0..15
        // (contiguous K halves, matching B).
        const __bf16* pa = sX + (rs * 16 + lhalf) * C + kBaseA;
        v8f c = {};
        #pragma unroll
        for (int s = 0; s < 4; ++s) {
            v8bf lo = *(const v8bf*)(pa + s * 32);
            v8bf hi = *(const v8bf*)(pa + s * 32 + 8);
            v16bf a = __builtin_shufflevector(lo, hi, 0, 1, 2, 3, 4, 5, 6, 7,
                                              8, 9, 10, 11, 12, 13, 14, 15);
            v16bf bw = (s == 0) ? bf0 : (s == 1) ? bf1 : (s == 2) ? bf2 : bf3;
            c = __builtin_amdgcn_wmma_f32_16x16x32_bf16(false, a, false, bw,
                                                        (short)0, c, false, false);
        }

        // C/D layout: lanes 0-15 rows v+0..7, lanes 16-31 rows v+8..15.
        int rBase = rowBlock + rs * 16 + ((lane >= 16) ? 8 : 0);
        float* hp = H + (long long)rBase * C + col;
        if (full) {
            #pragma unroll
            for (int v = 0; v < 8; ++v) hp[(long long)v * C] = c[v];
        } else {
            #pragma unroll
            for (int v = 0; v < 8; ++v)
                if (rBase + v < N) hp[(long long)v * C] = c[v];
        }
    }
}

// ---------------------------------------------------------------------------
// Launch
// ---------------------------------------------------------------------------
extern "C" void kernel_launch(void* const* d_in, const int* in_sizes, int n_in,
                              void* d_out, int out_size, void* d_ws, size_t ws_size,
                              hipStream_t stream) {
    const float* x  = (const float*)d_in[0];
    const int*   ei = (const int*)d_in[1];
    const float* W1 = (const float*)d_in[2];
    const float* b1 = (const float*)d_in[3];
    const float* W2 = (const float*)d_in[4];
    const float* b2 = (const float*)d_in[5];

    int N = in_sizes[0] / C;
    int E = in_sizes[1] / 2;
    const int* src = ei;
    const int* dst = ei + E;
    float* out = (float*)d_out;

    // Workspace layout: dinv [N] | h [N*C] | z [N*C]
    char* ws = (char*)d_ws;
    float* dinv = (float*)ws;
    size_t off  = ((size_t)N * sizeof(float) + 255) & ~(size_t)255;
    float* h    = (float*)(ws + off);
    size_t hb   = ((size_t)N * C * sizeof(float) + 255) & ~(size_t)255;
    float* z    = (float*)(ws + off + hb);

    long long NC = (long long)N * C;
    int gN  = (N + 255) / 256;
    int gE  = (E + 255) / 256;
    int gEw = (E + 7) / 8;                 // wave-per-edge, 8 waves/block
    int gNC = (int)((NC + 255) / 256);
    int gG  = (N + 63) / 64;               // 64 rows per GEMM block

    // Degrees (init 1.0 for self-loop) -> dinv
    k_fill<<<gN, 256, 0, stream>>>(dinv, 1.0f, N);
    k_degree<<<gE, 256, 0, stream>>>(dst, dinv, E);
    k_rsqrt_inplace<<<gN, 256, 0, stream>>>(dinv, N);

    // Layer 1: h = x@W1 ; z = scatter(h) ; z = relu(z + selfloop + b1)
    k_gemm_wmma<<<gG, 256, 0, stream>>>(x, W1, h, N);
    k_fill<<<gNC, 256, 0, stream>>>(z, 0.0f, NC);
    k_scatter<<<gEw, 256, 0, stream>>>(h, src, dst, dinv, z, E);
    k_finalize<<<gNC, 256, 0, stream>>>(z, h, dinv, b1, NC, 1);

    // Layer 2: h = z@W2 ; out = scatter(h) ; out = out + selfloop + b2
    k_gemm_wmma<<<gG, 256, 0, stream>>>(z, W2, h, N);
    k_fill<<<gNC, 256, 0, stream>>>(out, 0.0f, NC);
    k_scatter<<<gEw, 256, 0, stream>>>(h, src, dst, dinv, out, E);
    k_finalize<<<gNC, 256, 0, stream>>>(out, h, dinv, b2, NC, 0);
}
